// EncDecAttn_2293512536166
// MI455X (gfx1250) — matrix operand order repaired and checked by reading, not compiled, not run on hardware
//
#include <hip/hip_runtime.h>

typedef __attribute__((ext_vector_type(16))) _Float16 v16h;
typedef __attribute__((ext_vector_type(8)))  _Float16 v8h;
typedef __attribute__((ext_vector_type(8)))  float    v8f;

#define B_   8
#define NQ_  1024
#define NKV_ 2048
#define D_   512
#define H_   8
#define HD_  64
#define LOG2E 1.44269504088896340736f

// raw v_exp_f32: flush-to-zero on very negative args is exactly what softmax wants
#define EXP2(x) __builtin_amdgcn_exp2f(x)

// ---------- helpers ----------------------------------------------------------
// A-operand tile (16x32 f16) per CDNA5 ISA layout:
//   lane<16:  elems 0..7 -> K = 0..7,   elems 8..15 -> K = 16..23
//   lane>=16: elems 0..7 -> K = 8..15,  elems 8..15 -> K = 24..31
// caller passes p = rowptr + kbase + hi*8 ; halves live at +0 and +16.
__device__ __forceinline__ v16h load_a_tile(const _Float16* p) {
  union { v16h v; v8h h[2]; } u;
  u.h[0] = *(const v8h*)(p);
  u.h[1] = *(const v8h*)(p + 16);
  return u.v;
}

__device__ __forceinline__ v8f wmma_f16(v16h a, v16h b, v8f c) {
  return __builtin_amdgcn_wmma_f32_16x16x32_f16(false, a, false, b,
                                                (short)0, c, false, false);
}

// ---------- kernel 1: fp32 -> fp16 convert -----------------------------------
__global__ void cvt_f16(const float* __restrict__ s, _Float16* __restrict__ d, int n) {
  int i = blockIdx.x * blockDim.x + threadIdx.x;
  if (i < n) d[i] = (_Float16)s[i];
}

// ---------- kernel 2: C[M,512] = A[M,512] @ W^T ------------------------------
// One wave computes a 16(row) x 64(col) slab: A-operand reused across 4 WMMAs.
// MODE 0: f16 out, head-split  [b, h, row%Nper, d]
// MODE 2: f16 out, transposed  [b, h, d, row%Nper]   (for V^T)
// MODE 3: f32 out, flat        [row, col]            (final projection)
template<int MODE>
__global__ void gemm16(const _Float16* __restrict__ A,
                       const _Float16* __restrict__ W,
                       void* __restrict__ outp,
                       int M, int Nper) {
  const int lane = threadIdx.x & 31;
  const int hi   = lane >> 4;
  const int l15  = lane & 15;
  const int colGroups = D_ / 64;                       // 8
  const int row0     = (blockIdx.x / colGroups) * 16;
  const int colBase  = (blockIdx.x % colGroups) * 64;

  v8f acc[4] = {};
  const _Float16* arow = A + (size_t)(row0 + l15) * D_ + hi * 8;

  for (int kk = 0; kk < D_; kk += 32) {
    v16h a = load_a_tile(arow + kk);
#pragma unroll
    for (int t = 0; t < 4; ++t) {
      // B-operand: lane n holds W[col][kk + hi*8 + 0..7] and W[col][kk + 16 + hi*8 + 0..7] (table order)
      const _Float16* wp = W + (size_t)(colBase + t * 16 + l15) * D_ + kk + hi * 8;
      v16h b = load_a_tile(wp);
      acc[t] = wmma_f16(a, b, acc[t]);
    }
  }

  if (MODE == 3) {
    float* O = (float*)outp;
#pragma unroll
    for (int t = 0; t < 4; ++t) {
      int col = colBase + t * 16 + l15;
#pragma unroll
      for (int r = 0; r < 8; ++r)
        O[(size_t)(row0 + r + 8 * hi) * D_ + col] = acc[t][r];
    }
  } else if (MODE == 2) {
    _Float16* O = (_Float16*)outp;
    int b   = row0 / Nper;
    int rr0 = row0 % Nper;
#pragma unroll
    for (int t = 0; t < 4; ++t) {
      int col = colBase + t * 16 + l15;
      int h = col >> 6, d = col & 63;
      union { v8h v; _Float16 e[8]; } s;
#pragma unroll
      for (int r = 0; r < 8; ++r) s.e[r] = (_Float16)acc[t][r];
      *(v8h*)(O + (((size_t)(b * H_ + h) * HD_ + d) * Nper + rr0 + 8 * hi)) = s.v;
    }
  } else {
    _Float16* O = (_Float16*)outp;
    int b   = row0 / Nper;
    int rr0 = row0 % Nper;
#pragma unroll
    for (int t = 0; t < 4; ++t) {
      int col = colBase + t * 16 + l15;
      int h = col >> 6, d = col & 63;
#pragma unroll
      for (int r = 0; r < 8; ++r)
        O[(((size_t)(b * H_ + h) * Nper + rr0 + r + 8 * hi) * HD_) + d] = (_Float16)acc[t][r];
    }
  }
}

// ---------- kernel 3: flash attention ----------------------------------------
// 128-thread workgroup = 4 waves covering 64 queries of one (b,h).
// Each 32-key chunk of K (32x64 f16) and V^T (64x32 f16) is staged into LDS
// once (one 16B global load + ds_store per thread per array) and consumed by
// all 4 waves -> 4x less L2 traffic than per-wave streaming, with the
// register-light 16-query flash loop per wave (no spills).
// Per wave: S^T = K @ Q^T with pos_bias as the WMMA C operand; online softmax
// with per-column stats (column == lane pair); O^T += V^T @ P^T.
__global__ void attn_lds(const _Float16* __restrict__ Qh,   // [B,H,NQ,64]
                         const _Float16* __restrict__ Kh,   // [B,H,NKV,64]
                         const _Float16* __restrict__ Vt,   // [B,H,64,NKV]
                         const float*    __restrict__ bias, // [1,H,NQ,NKV]
                         _Float16*       __restrict__ AO) { // [B,NQ,512] f16
  __shared__ __align__(32) _Float16 Ks[32 * 64];   // [key][d]   4 KB
  __shared__ __align__(32) _Float16 Vs[64 * 32];   // [d][key]   4 KB

  const int tid  = threadIdx.x;
  const int wave = tid >> 5;
  const int lane = tid & 31;
  const int hi   = lane >> 4;
  const int l15  = lane & 15;

  const int qBlocks = NQ_ / 64;                    // 16
  const int qb = blockIdx.x % qBlocks;
  const int bh = blockIdx.x / qBlocks;
  const int h  = bh % H_;
  const int b  = bh / H_;
  const int q0 = qb * 64 + wave * 16;              // this wave's 16 queries

  const _Float16* Qb = Qh + ((size_t)bh * NQ_ + q0) * HD_;
  const _Float16* Kb = Kh + (size_t)bh * NKV_ * HD_;
  const _Float16* Vb = Vt + (size_t)bh * HD_ * NKV_;
  const float*   bp0 = bias + ((size_t)h * NQ_ + q0 + l15) * NKV_ + hi * 8;

  // Q as B-operand (columns = queries): table order, two 16B halves per lane.
  v16h bq0 = load_a_tile(Qb + (size_t)l15 * HD_ + hi * 8);
  v16h bq1 = load_a_tile(Qb + (size_t)l15 * HD_ + 32 + hi * 8);

  // cooperative staging indices: one 16B (v8h) segment per thread per array
  const int krow = tid >> 2;                // 0..31  (key row)
  const int kseg = (tid & 3) * 16;          // 0,16,32,48 (d offset)
  const int vrow = tid >> 1;                // 0..63  (d row)
  const int vseg = (tid & 1) * 16;          // 0,16   (key offset)

  v8f o0 = {}, o1 = {}, o2 = {}, o3 = {};
  float m = -3.0e38f, lsum = 0.f;

  for (int kb = 0; kb < NKV_; kb += 32) {
    // --- stage K / V^T chunk into LDS
    *(v8h*)(&Ks[krow * 64 + kseg]) =
        *(const v8h*)(Kb + (size_t)(kb + krow) * HD_ + kseg);
    *(v8h*)(&Vs[vrow * 32 + vseg]) =
        *(const v8h*)(Vb + (size_t)vrow * NKV_ + kb + vseg);
    __syncthreads();

    // --- bias tiles as C-operand init (C layout: row = key r+8*hi, col = l15)
    v8f st0, st1;
    {
      const float* bp = bp0 + kb;
      float4 f0 = *(const float4*)(bp);
      float4 f1 = *(const float4*)(bp + 4);
      st0[0]=f0.x; st0[1]=f0.y; st0[2]=f0.z; st0[3]=f0.w;
      st0[4]=f1.x; st0[5]=f1.y; st0[6]=f1.z; st0[7]=f1.w;
      float4 g0 = *(const float4*)(bp + 16);
      float4 g1 = *(const float4*)(bp + 20);
      st1[0]=g0.x; st1[1]=g0.y; st1[2]=g0.z; st1[3]=g0.w;
      st1[4]=g1.x; st1[5]=g1.y; st1[6]=g1.z; st1[7]=g1.w;
    }
    // --- S^T = K @ Q^T + bias  (K A-tiles from LDS)
    const _Float16* krow0 = &Ks[l15 * 64 + hi * 8];
    const _Float16* krow1 = &Ks[(16 + l15) * 64 + hi * 8];
    st0 = wmma_f16(load_a_tile(krow0),      bq0, st0);
    st0 = wmma_f16(load_a_tile(krow0 + 32), bq1, st0);
    st1 = wmma_f16(load_a_tile(krow1),      bq0, st1);
    st1 = wmma_f16(load_a_tile(krow1 + 32), bq1, st1);

    // --- online softmax over keys (per query column)
    float cmax = st0[0];
#pragma unroll
    for (int r = 1; r < 8; ++r) cmax = fmaxf(cmax, st0[r]);
#pragma unroll
    for (int r = 0; r < 8; ++r) cmax = fmaxf(cmax, st1[r]);
    cmax = fmaxf(cmax, __shfl_xor(cmax, 16, 32));
    float mnew = fmaxf(m, cmax);
    float sc = EXP2((m - mnew) * LOG2E);

    float p0[8], p1[8], cs = 0.f;
#pragma unroll
    for (int r = 0; r < 8; ++r) {
      p0[r] = EXP2((st0[r] - mnew) * LOG2E);
      p1[r] = EXP2((st1[r] - mnew) * LOG2E);
      cs += p0[r] + p1[r];
    }
    cs += __shfl_xor(cs, 16, 32);
    lsum = lsum * sc + cs;
    m = mnew;
#pragma unroll
    for (int r = 0; r < 8; ++r) { o0[r]*=sc; o1[r]*=sc; o2[r]*=sc; o3[r]*=sc; }

    // --- P^T (C layout) -> B-operand (rows = 32 keys, cols = q), table order:
    //     elems 0..7 = keys 8*hi + r (own p0), elems 8..15 = keys 16 + 8*hi + r (own p1).
    v16h pb;
#pragma unroll
    for (int r = 0; r < 8; ++r) {
      pb[r]     = (_Float16)p0[r];
      pb[r + 8] = (_Float16)p1[r];
    }

    // --- O^T += V^T @ P^T  (V^T A-tiles from LDS)
    const _Float16* v0 = &Vs[l15 * 32 + hi * 8];
    o0 = wmma_f16(load_a_tile(v0),            pb, o0);
    o1 = wmma_f16(load_a_tile(v0 + 16 * 32),  pb, o1);
    o2 = wmma_f16(load_a_tile(v0 + 32 * 32),  pb, o2);
    o3 = wmma_f16(load_a_tile(v0 + 48 * 32),  pb, o3);
    __syncthreads();   // protect LDS before next chunk is staged
  }

  // --- normalize and store as [b, q, h*64 + d] f16 (8 contiguous halfs/tile)
  float inv = 1.0f / lsum;
  _Float16* outq = AO + ((size_t)(b * NQ_ + q0 + l15)) * D_ + h * HD_ + hi * 8;
  const v8f* ot[4] = { &o0, &o1, &o2, &o3 };
#pragma unroll
  for (int t = 0; t < 4; ++t) {
    union { v8h v; _Float16 e[8]; } s;
#pragma unroll
    for (int r = 0; r < 8; ++r) s.e[r] = (_Float16)((*ot[t])[r] * inv);
    *(v8h*)(outq + t * 16) = s.v;
  }
}

// ---------- host launch ------------------------------------------------------
extern "C" void kernel_launch(void* const* d_in, const int* in_sizes, int n_in,
                              void* d_out, int out_size, void* d_ws, size_t ws_size,
                              hipStream_t stream) {
  (void)in_sizes; (void)n_in; (void)out_size; (void)ws_size;
  const float* x    = (const float*)d_in[0];
  const float* enc  = (const float*)d_in[1];
  const float* bias = (const float*)d_in[2];
  const float* Wq   = (const float*)d_in[3];
  const float* Wk   = (const float*)d_in[4];
  const float* Wv   = (const float*)d_in[5];
  const float* Wo   = (const float*)d_in[6];

  _Float16* ws = (_Float16*)d_ws;
  const size_t nX  = (size_t)B_ * NQ_ * D_;    // 4,194,304
  const size_t nE  = (size_t)B_ * NKV_ * D_;   // 8,388,608
  const size_t nW  = (size_t)D_ * D_;          //   262,144
  _Float16* xh   = ws;
  _Float16* ench = xh + nX;
  _Float16* wqh  = ench + nE;
  _Float16* wkh  = wqh + nW;
  _Float16* wvh  = wkh + nW;
  _Float16* woh  = wvh + nW;
  _Float16* qh   = woh + nW;        // [B,H,NQ,64]
  _Float16* kh   = qh + nX;         // [B,H,NKV,64]
  _Float16* vth  = kh + nE;         // [B,H,64,NKV]
  _Float16* ath  = vth + nE;        // [B,NQ,512]

  cvt_f16<<<(nX + 255) / 256, 256, 0, stream>>>(x,   xh,   (int)nX);
  cvt_f16<<<(nE + 255) / 256, 256, 0, stream>>>(enc, ench, (int)nE);
  cvt_f16<<<(nW + 255) / 256, 256, 0, stream>>>(Wq,  wqh,  (int)nW);
  cvt_f16<<<(nW + 255) / 256, 256, 0, stream>>>(Wk,  wkh,  (int)nW);
  cvt_f16<<<(nW + 255) / 256, 256, 0, stream>>>(Wv,  wvh,  (int)nW);
  cvt_f16<<<(nW + 255) / 256, 256, 0, stream>>>(Wo,  woh,  (int)nW);

  const int Mq = B_ * NQ_;    // 8192
  const int Mk = B_ * NKV_;   // 16384
  gemm16<0><<<(Mq / 16) * (D_ / 64), 32, 0, stream>>>(xh,   wqh, (void*)qh,  Mq, NQ_);
  gemm16<0><<<(Mk / 16) * (D_ / 64), 32, 0, stream>>>(ench, wkh, (void*)kh,  Mk, NKV_);
  gemm16<2><<<(Mk / 16) * (D_ / 64), 32, 0, stream>>>(ench, wvh, (void*)vth, Mk, NKV_);

  attn_lds<<<B_ * H_ * (NQ_ / 64), 128, 0, stream>>>(qh, kh, vth, bias, ath);

  gemm16<3><<<(Mq / 16) * (D_ / 64), 32, 0, stream>>>(ath, woh, d_out, Mq, NQ_);
}
